// MambaBlock_39290360824602
// MI455X (gfx1250) — matrix operand order repaired and checked by reading, not compiled, not run on hardware
//
#include <hip/hip_runtime.h>
#include <math.h>

// ---------------- Problem constants (match reference) ----------------
#define DIM_C   1024
#define DIN_C   2048
#define NST     16      // N state dim
#define NG_C    32
#define DTR     64      // DT_RANK
#define DCONV   4
#define B_C     2
#define L_C     1024
#define PG_C    (L_C / NG_C)     // 32
#define M_C     (B_C * L_C)      // 2048
#define XDBL_N  (DTR + 2 * NST)  // 96
#define G_C     (NG_C - 1)       // 31

typedef __attribute__((ext_vector_type(16))) __bf16 v16bf;
typedef __attribute__((ext_vector_type(8)))  float  v8f;
typedef __attribute__((ext_vector_type(4)))  unsigned int v4u;
typedef __attribute__((ext_vector_type(4)))  int    v4i;
typedef __attribute__((ext_vector_type(8)))  int    v8i;

union V16U { v16bf v; int4 q[2]; };

#if __has_builtin(__builtin_amdgcn_tensor_load_to_lds)
#define HAVE_TDM 1
#else
#define HAVE_TDM 0
#endif

// ---------------------------------------------------------------------
// TDM: async DMA of a 2-D tile (tile_d0 x tile_d1, 2-byte elements) from a
// row-major tensor (row stride = stride0 elements) into LDS at byte offset
// ldsOff.  D# packing per CDNA5 ISA 8.3/8.4 (count=1, type=2, mask=0).
__device__ __forceinline__ void tdm_load_tile_bf16(unsigned ldsOff, const void* gaddr,
                                                   int tensor_d0, int tensor_d1,
                                                   int tile_d0, int tile_d1,
                                                   int stride0) {
#if HAVE_TDM
    unsigned long long ga = (unsigned long long)gaddr;
    v4u g0;
    g0[0] = 1u;                                        // count=1, user desc
    g0[1] = ldsOff;                                    // lds_addr (bytes)
    g0[2] = (unsigned)(ga & 0xffffffffu);              // global_addr[31:0]
    g0[3] = (unsigned)((ga >> 32) & 0x01ffffffu)       // global_addr[56:32]
          | (2u << 30);                                // type=2 ("image")
    v8i g1;
    g1[0] = 0x10000;                                   // workgroup_mask=0, data_size=1(2B)
    g1[1] = (int)(((unsigned)tensor_d0 & 0xffffu) << 16);            // dim0[15:0]
    g1[2] = (int)((((unsigned)tensor_d0 >> 16) & 0xffffu)
          | (((unsigned)tensor_d1 & 0xffffu) << 16));                // dim0[31:16], dim1[15:0]
    g1[3] = (int)((((unsigned)tensor_d1 >> 16) & 0xffffu)
          | ((unsigned)tile_d0 << 16));                              // dim1[31:16], tile_dim0
    g1[4] = tile_d1;                                   // tile_dim1 (tile_dim2=0)
    g1[5] = stride0;                                   // tensor_dim0_stride[31:0]
    g1[6] = 0;                                         // stride0[47:32], stride1[15:0]
    g1[7] = 0;                                         // stride1[47:16]
    v4i z4 = {0, 0, 0, 0};
#if __clang_major__ >= 23
    v8i z8 = {0, 0, 0, 0, 0, 0, 0, 0};
    __builtin_amdgcn_tensor_load_to_lds(g0, g1, z4, z4, z8, 0);
#else
    __builtin_amdgcn_tensor_load_to_lds(g0, g1, z4, z4, 0);
#endif
#else
    (void)ldsOff; (void)gaddr; (void)tensor_d0; (void)tensor_d1;
    (void)tile_d0; (void)tile_d1; (void)stride0;
#endif
}

template <int Nw>
__device__ __forceinline__ void wait_tensorcnt() {
#if __has_builtin(__builtin_amdgcn_s_wait_tensorcnt)
    __builtin_amdgcn_s_wait_tensorcnt((short)Nw);
#else
    if (Nw == 0) asm volatile("s_wait_tensorcnt 0x0" ::: "memory");
    else         asm volatile("s_wait_tensorcnt 0x1" ::: "memory");
#endif
}

// ---------------------------------------------------------------------
// bf16 fragment loaders (v_wmma_f32_16x16x32_bf16, wave32).
// A 16x32 (MxK): lane l holds row m=l%16; elems j=0..7 -> K=halfK8+j,
// j=8..15 -> K=16+halfK8+(j-8), halfK8=(l>=16)?8:0  => two 16B chunks.
__device__ __forceinline__ v16bf frag_a_lds(const __bf16* As, int lane, int mi) {
    const __bf16* p = As + ((lane & 15) + mi) * 32 + ((lane & 16) ? 8 : 0);
    V16U t;
    t.q[0] = *(const int4*)(p);
    t.q[1] = *(const int4*)(p + 16);
    return t.v;
}

// B 32x16 (KxN): lane l holds column n=l%16; elems j=0..7 -> K=halfK8+j,
// j=8..15 -> K=16+halfK8+(j-8), halfK8=(l>=16)?8:0  => two 16B chunks from W[n][k0..].
__device__ __forceinline__ v16bf load_frag_b_bf(const __bf16* __restrict__ base,
                                                int ldb, int nrow, int k0, int lane) {
    const __bf16* p = base + (size_t)nrow * ldb + k0 + ((lane & 16) ? 8 : 0);
    V16U t;
    t.q[0] = *(const int4*)(p);
    t.q[1] = *(const int4*)(p + 16);
    return t.v;
}

__device__ __forceinline__ float softplus_f(float x) {
    return (x > 20.f) ? x : log1pf(__expf(x));
}

// ---------------------------------------------------------------------
// f32 -> bf16 conversion pass
__global__ void cvt_bf16_kernel(const float* __restrict__ src,
                                __bf16* __restrict__ dst, int n) {
    int i = blockIdx.x * blockDim.x + threadIdx.x;
    if (i < n) dst[i] = (__bf16)src[i];
}

// ---------------------------------------------------------------------
// GEMM: C[M,N] = epilogue( A[M,K(lda)] @ B[N,K(ldb)]^T ), A/B bf16, C f32.
// Block = 128 threads (4 waves); block tile 32(M) x 256(N); wave strip 32x64
// (8 wmma / K-step). A tile staged in LDS by TDM (double-buffered) and shared
// by all 4 waves.  EPI: 0=none, 1=softplus(x+bias[n]).  GUARD: N-tile bounds.
template <int EPI, bool GUARD>
__global__ void wmma_gemm(const __bf16* __restrict__ A, const __bf16* __restrict__ Bw,
                          const float* __restrict__ bias, float* __restrict__ C,
                          int M, int N, int K, int lda, int ldb) {
    __shared__ __bf16 As[2][32 * 32];        // 2 x 2KB double buffer
    const int lane   = threadIdx.x & 31;
    const int wave   = threadIdx.x >> 5;
    const int tileM  = blockIdx.x * 32;
    const int tileN0 = (blockIdx.y * 4 + wave) * 64;

    v8f acc[2][4];
#pragma unroll
    for (int i = 0; i < 2; ++i)
#pragma unroll
        for (int t = 0; t < 4; ++t) acc[i][t] = v8f{};

    const int nl = lane & 15;

#if HAVE_TDM
    if (wave == 0) {
        tdm_load_tile_bf16(0u, A + (size_t)tileM * lda, lda, M, 32, 32, lda);
    }
    int ib = 0;
    for (int k0 = 0; k0 < K; k0 += 32) {
        if (wave == 0) {
            if (k0 + 32 < K) {   // issue next tile DMA, then wait for current
                tdm_load_tile_bf16((unsigned)((ib ^ 1) * 2048),
                                   A + (size_t)tileM * lda + (k0 + 32),
                                   lda, M, 32, 32, lda);
                wait_tensorcnt<1>();
            } else {
                wait_tensorcnt<0>();
            }
        }
        __syncthreads();
        const __bf16* as = &As[ib][0];
        v16bf a0 = frag_a_lds(as, lane, 0);
        v16bf a1 = frag_a_lds(as, lane, 16);
#pragma unroll
        for (int t = 0; t < 4; ++t) {
            int nt = tileN0 + t * 16;
            if (GUARD && nt >= N) continue;
            if (k0 + 32 < K) {
                __builtin_prefetch(Bw + (size_t)(nt + nl) * ldb + k0 + 32, 0, 0);
            }
            v16bf b = load_frag_b_bf(Bw, ldb, nt + nl, k0, lane);
            acc[0][t] = __builtin_amdgcn_wmma_f32_16x16x32_bf16(
                false, a0, false, b, (short)0, acc[0][t], false, false);
            acc[1][t] = __builtin_amdgcn_wmma_f32_16x16x32_bf16(
                false, a1, false, b, (short)0, acc[1][t], false, false);
        }
        __syncthreads();
        ib ^= 1;
    }
#else
    const int crow = threadIdx.x >> 2;        // 0..31 : A row copied by thread
    const int ccol = (threadIdx.x & 3) * 8;   // 0,8,16,24
    for (int k0 = 0; k0 < K; k0 += 32) {
        if (k0 + 32 < K) {
            __builtin_prefetch(A + (size_t)(tileM + crow) * lda + (k0 + 32) + ccol, 0, 0);
        }
        __syncthreads();
        *(int4*)(&As[0][0] + crow * 32 + ccol) =
            *(const int4*)(A + (size_t)(tileM + crow) * lda + k0 + ccol);
        __syncthreads();
        v16bf a0 = frag_a_lds(&As[0][0], lane, 0);
        v16bf a1 = frag_a_lds(&As[0][0], lane, 16);
#pragma unroll
        for (int t = 0; t < 4; ++t) {
            int nt = tileN0 + t * 16;
            if (GUARD && nt >= N) continue;
            v16bf b = load_frag_b_bf(Bw, ldb, nt + nl, k0, lane);
            acc[0][t] = __builtin_amdgcn_wmma_f32_16x16x32_bf16(
                false, a0, false, b, (short)0, acc[0][t], false, false);
            acc[1][t] = __builtin_amdgcn_wmma_f32_16x16x32_bf16(
                false, a1, false, b, (short)0, acc[1][t], false, false);
        }
    }
#endif

    // C/D layout: VGPR r -> row (+r, +8 for lanes>=16); col = lane%16
    const int rowoff = (lane & 16) ? 8 : 0;
#pragma unroll
    for (int mi = 0; mi < 2; ++mi) {
#pragma unroll
        for (int t = 0; t < 4; ++t) {
            int nt = tileN0 + t * 16;
            if (GUARD && nt >= N) continue;
            float* cp = C + (size_t)(tileM + mi * 16 + rowoff) * N + nt + nl;
#pragma unroll
            for (int r = 0; r < 8; ++r) {
                float v = acc[mi][t][r];
                if (EPI == 1) v = softplus_f(v + bias[nt + nl]);
                cp[(size_t)r * N] = v;
            }
        }
    }
}

// ---------------------------------------------------------------------
// Depthwise causal conv (width 4) + bias + SiLU. Writes f32 (scan) + bf16 (GEMM).
__global__ void conv_silu_kernel(const float* __restrict__ xr,
                                 const float* __restrict__ conv_w,
                                 const float* __restrict__ conv_b,
                                 float* __restrict__ u,
                                 __bf16* __restrict__ u_bf) {
    int idx = blockIdx.x * blockDim.x + threadIdx.x;
    if (idx >= M_C * DIN_C) return;
    int d = idx & (DIN_C - 1);
    int m = idx >> 11;
    int l = m & (L_C - 1);
    int b = m >> 10;
    float acc = conv_b[d];
#pragma unroll
    for (int i = 0; i < DCONV; ++i) {
        int ll = l - (DCONV - 1) + i;
        if (ll >= 0) {
            acc += xr[(size_t)(b * L_C + ll) * (2 * DIN_C) + d] * conv_w[d * DCONV + i];
        }
    }
    float s = 1.f / (1.f + __expf(-acc));
    float v = acc * s;
    u[(size_t)m * DIN_C + d] = v;
    u_bf[(size_t)m * DIN_C + d] = (__bf16)v;
}

// ---------------------------------------------------------------------
// Selective scan per (b, g, d); fuses +u*D and y*silu(res); emits bf16 yg.
__global__ void scan_kernel(const float* __restrict__ delta,
                            const float* __restrict__ u,
                            const float* __restrict__ xdbl,
                            const float* __restrict__ xr,
                            const float* __restrict__ latent,
                            const float* __restrict__ A_log,
                            const float* __restrict__ Dp,
                            __bf16* __restrict__ yg) {
    int idx = blockIdx.x * blockDim.x + threadIdx.x;   // (b*NG+g)*DIN + d
    if (idx >= B_C * NG_C * DIN_C) return;
    int d  = idx & (DIN_C - 1);
    int bg = idx >> 11;
    int mbase = bg * PG_C;                              // = b*L + g*PG

    float Arow[NST];
#pragma unroll
    for (int n = 0; n < NST; ++n) Arow[n] = -__expf(A_log[(size_t)d * NST + n]);

    float h[NST];
#pragma unroll
    for (int n = 0; n < NST; ++n) h[n] = 0.f;

    const float Dd = Dp[d];
    const float* latp = latent + ((size_t)bg * DIN_C + d) * NST;

    for (int p = 0; p < PG_C; ++p) {
        int m = mbase + p;
        float dv = delta[(size_t)m * DIN_C + d];
        float uv = u[(size_t)m * DIN_C + d];
        const float* bc = xdbl + (size_t)m * XDBL_N + DTR;   // B[0..15], C[16..31]
        float y = 0.f;
#pragma unroll
        for (int n = 0; n < NST; ++n) {
            float dA  = __expf(dv * Arow[n]);
            float dBu = dv * bc[n] * uv;
            if (p == 0) dBu += latp[n];
            h[n] = dA * h[n] + dBu;
            y += h[n] * bc[NST + n];
        }
        y += uv * Dd;
        float r   = xr[(size_t)m * (2 * DIN_C) + DIN_C + d];
        float sig = 1.f / (1.f + __expf(-r));
        yg[(size_t)m * DIN_C + d] = (__bf16)(y * (r * sig));
    }
}

// ---------------------------------------------------------------------
// qkv = latent[:, :G] @ W_cattn^T   (per (b,g,d): 16 -> 48 matvec)
__global__ void qkv_kernel(const float* __restrict__ latent,
                           const float* __restrict__ Wc,
                           float* __restrict__ qkv) {
    int idx = blockIdx.x * blockDim.x + threadIdx.x;   // (b*G+g)*DIN + d
    if (idx >= B_C * G_C * DIN_C) return;
    int d  = idx & (DIN_C - 1);
    int bg = idx >> 11;                                 // b*G + g
    int g  = bg % G_C;
    int b  = bg / G_C;
    const float* lp = latent + (((size_t)(b * NG_C + g) * DIN_C) + d) * NST;
    float lat[NST];
#pragma unroll
    for (int n = 0; n < NST; ++n) lat[n] = lp[n];
    float* op = qkv + (size_t)idx * (3 * NST);
#pragma unroll
    for (int j = 0; j < 3 * NST; ++j) {
        float s = 0.f;
#pragma unroll
        for (int n = 0; n < NST; ++n) s += Wc[j * NST + n] * lat[n];
        op[j] = s;
    }
}

// scores[b,i,j] = scale * q_i . k_j   (dot over DIN*NST features)
__global__ void scores_kernel(const float* __restrict__ qkv, float* __restrict__ scores) {
    int bij = blockIdx.x;
    int j = bij % G_C;
    int t = bij / G_C;
    int i = t % G_C;
    int b = t / G_C;
    __shared__ float red[256];
    const float* qb = qkv + ((size_t)(b * G_C + i) * DIN_C) * (3 * NST);
    const float* kb = qkv + ((size_t)(b * G_C + j) * DIN_C) * (3 * NST) + NST;
    float s = 0.f;
    for (int d = threadIdx.x; d < DIN_C; d += 256) {
        const float* qp = qb + (size_t)d * (3 * NST);
        const float* kp = kb + (size_t)d * (3 * NST);
#pragma unroll
        for (int n = 0; n < NST; ++n) s += qp[n] * kp[n];
    }
    red[threadIdx.x] = s;
    __syncthreads();
    for (int off = 128; off > 0; off >>= 1) {
        if (threadIdx.x < off) red[threadIdx.x] += red[threadIdx.x + off];
        __syncthreads();
    }
    if (threadIdx.x == 0) {
        const float scale = 1.f / sqrtf((float)(DIN_C * NST));
        scores[(size_t)(b * G_C + i) * G_C + j] = red[0] * scale;
    }
}

// causal softmax over one row of scores (wave32)
__global__ void softmax_kernel(float* __restrict__ scores) {
    int row = blockIdx.x;            // b*G + i
    int i = row % G_C;
    int lane = threadIdx.x;          // 0..31
    bool act = (lane <= i) && (lane < G_C);
    float v = act ? scores[(size_t)row * G_C + lane] : -__builtin_inff();
    float mx = v;
    for (int off = 16; off > 0; off >>= 1) mx = fmaxf(mx, __shfl_xor(mx, off, 32));
    float e = act ? __expf(v - mx) : 0.f;
    float sm = e;
    for (int off = 16; off > 0; off >>= 1) sm += __shfl_xor(sm, off, 32);
    if (lane < G_C) scores[(size_t)row * G_C + lane] = act ? (e / sm) : 0.f;
}

// lat_out[b,g,d,n]: g==0 -> 0 ; g>0 -> sum_j attn[b,g-1,j] * v[b,j,d,n]
__global__ void latout_kernel(const float* __restrict__ qkv,
                              const float* __restrict__ attn,
                              float* __restrict__ lat_out) {
    int idx = blockIdx.x * blockDim.x + threadIdx.x;
    if (idx >= B_C * NG_C * DIN_C * NST) return;
    int n = idx & (NST - 1);
    int r = idx >> 4;
    int d = r & (DIN_C - 1);
    int r2 = r >> 11;
    int g = r2 & (NG_C - 1);
    int b = r2 >> 5;
    float v = 0.f;
    if (g > 0) {
        int i = g - 1;
        const float* ar = attn + (size_t)(b * G_C + i) * G_C;
        for (int j = 0; j <= i; ++j) {
            v += ar[j] * qkv[(((size_t)(b * G_C + j) * DIN_C) + d) * (3 * NST) + 2 * NST + n];
        }
    }
    lat_out[idx] = v;
}

// ---------------------------------------------------------------------
extern "C" void kernel_launch(void* const* d_in, const int* in_sizes, int n_in,
                              void* d_out, int out_size, void* d_ws, size_t ws_size,
                              hipStream_t stream) {
    const float* x      = (const float*)d_in[0];
    const float* latent = (const float*)d_in[1];
    const float* W_in   = (const float*)d_in[2];
    const float* conv_w = (const float*)d_in[3];
    const float* conv_b = (const float*)d_in[4];
    const float* W_x    = (const float*)d_in[5];
    const float* W_dt   = (const float*)d_in[6];
    const float* b_dt   = (const float*)d_in[7];
    const float* W_out  = (const float*)d_in[8];
    const float* A_log  = (const float*)d_in[9];
    const float* Dvec   = (const float*)d_in[10];
    const float* W_catt = (const float*)d_in[11];

    float* out     = (float*)d_out;                       // (B, L, DIM)
    float* lat_out = out + (size_t)B_C * L_C * DIM_C;     // (B, NG, DIN, NST)

    // ---- workspace layout (f32 region, then bf16 region) ----
    float* ws    = (float*)d_ws;
    float* xr    = ws;                                    // 2048 x 4096
    float* u     = xr    + (size_t)M_C * (2 * DIN_C);     // 2048 x 2048
    float* xdbl  = u     + (size_t)M_C * DIN_C;           // 2048 x 96
    float* delta = xdbl  + (size_t)M_C * XDBL_N;          // 2048 x 2048
    float* qkv   = delta + (size_t)M_C * DIN_C;           // (B*G*DIN) x 48
    float* sc    = qkv   + (size_t)B_C * G_C * DIN_C * (3 * NST);
    __bf16* x_bf    = (__bf16*)(sc + 2048);
    __bf16* Win_bf  = x_bf    + (size_t)M_C * DIM_C;          // 4096 x 1024
    __bf16* Wx_bf   = Win_bf  + (size_t)2 * DIN_C * DIM_C;    // 96 x 2048
    __bf16* Wdt_bf  = Wx_bf   + (size_t)XDBL_N * DIN_C;       // 2048 x 64
    __bf16* Wout_bf = Wdt_bf  + (size_t)DIN_C * DTR;          // 1024 x 2048
    __bf16* u_bf    = Wout_bf + (size_t)DIM_C * DIN_C;        // 2048 x 2048
    __bf16* xdbl_bf = u_bf    + (size_t)M_C * DIN_C;          // 2048 x 96
    __bf16* yg_bf   = xdbl_bf + (size_t)M_C * XDBL_N;         // 2048 x 2048

    dim3 blk(128, 1, 1);
    auto cvt = [&](const float* s, __bf16* dst, int n) {
        cvt_bf16_kernel<<<(n + 255) / 256, 256, 0, stream>>>(s, dst, n);
    };

    // 0) one-time bf16 conversions
    cvt(x,     x_bf,    M_C * DIM_C);
    cvt(W_in,  Win_bf,  2 * DIN_C * DIM_C);
    cvt(W_x,   Wx_bf,   XDBL_N * DIN_C);
    cvt(W_dt,  Wdt_bf,  DIN_C * DTR);
    cvt(W_out, Wout_bf, DIM_C * DIN_C);

    // 1) xr = x @ W_in^T            (M=2048, N=4096, K=1024)
    wmma_gemm<0, false><<<dim3(M_C / 32, 16), blk, 0, stream>>>(
        x_bf, Win_bf, nullptr, xr, M_C, 2 * DIN_C, DIM_C, DIM_C, DIM_C);

    // 2) depthwise conv + silu -> u (f32 + bf16)
    conv_silu_kernel<<<(M_C * DIN_C + 255) / 256, 256, 0, stream>>>(
        xr, conv_w, conv_b, u, u_bf);

    // 3) x_dbl = u @ W_x^T          (M=2048, N=96, K=2048)
    wmma_gemm<0, true><<<dim3(M_C / 32, 1), blk, 0, stream>>>(
        u_bf, Wx_bf, nullptr, xdbl, M_C, XDBL_N, DIN_C, DIN_C, DIN_C);
    cvt(xdbl, xdbl_bf, M_C * XDBL_N);

    // 4) delta = softplus(x_dbl[:, :64] @ W_dt^T + b_dt)   (N=2048, K=64)
    wmma_gemm<1, false><<<dim3(M_C / 32, 8), blk, 0, stream>>>(
        xdbl_bf, Wdt_bf, b_dt, delta, M_C, DIN_C, DTR, XDBL_N, DTR);

    // 5) selective scan + gating -> yg (bf16)
    scan_kernel<<<(B_C * NG_C * DIN_C + 255) / 256, 256, 0, stream>>>(
        delta, u, xdbl, xr, latent, A_log, Dvec, yg_bf);

    // 6) out = yg @ W_out^T         (M=2048, N=1024, K=2048)
    wmma_gemm<0, false><<<dim3(M_C / 32, 4), blk, 0, stream>>>(
        yg_bf, Wout_bf, nullptr, out, M_C, DIM_C, DIN_C, DIN_C, DIN_C);

    // 7) latent cross-attention -> lat_out
    qkv_kernel<<<(B_C * G_C * DIN_C + 255) / 256, 256, 0, stream>>>(latent, W_catt, qkv);
    scores_kernel<<<B_C * G_C * G_C, 256, 0, stream>>>(qkv, sc);
    softmax_kernel<<<B_C * G_C, 32, 0, stream>>>(sc);
    latout_kernel<<<(B_C * NG_C * DIN_C * NST + 255) / 256, 256, 0, stream>>>(qkv, sc, lat_out);
}
